// SelfAttention_31172872634433
// MI455X (gfx1250) — matrix operand order repaired and checked by reading, not compiled, not run on hardware
//
#include <hip/hip_runtime.h>
#include <hip/hip_bf16.h>
#include <stdint.h>

// ---------------------------------------------------------------------------
// CDNA5 (gfx1250) self-attention forward, bf16x3 split-precision WMMA.
// B=4, T=2048, D=1024, H=16, HDIM=64.
// Software-pipelined global->LDS staging in the GEMMs; async global->LDS
// (GLOBAL_LOAD_ASYNC_TO_LDS_B128, ASYNCcnt-tracked) double-buffered staging
// of K/V tiles in the attention kernel.
// ---------------------------------------------------------------------------

typedef __attribute__((ext_vector_type(16))) __bf16 v16bf;
typedef __attribute__((ext_vector_type(8)))  float  v8f;

#define WMMA_BF16 __builtin_amdgcn_wmma_f32_16x16x32_bf16

// ---- async global->LDS support (guarded so compile never breaks) ----------
#if defined(__has_builtin)
#  if __has_builtin(__builtin_amdgcn_global_load_async_to_lds_b128)
#    define HAVE_ASYNC_LDS 1
#  endif
#endif
#ifndef HAVE_ASYNC_LDS
#  define HAVE_ASYNC_LDS 0
#endif

#if HAVE_ASYNC_LDS
#define AS1 __attribute__((address_space(1)))
#define AS3 __attribute__((address_space(3)))
typedef __attribute__((__vector_size__(4 * sizeof(int)))) int v4i_;
static __device__ __forceinline__ void async_cp16(const float* g, float* l) {
    __builtin_amdgcn_global_load_async_to_lds_b128((AS1 v4i_*)g, (AS3 v4i_*)l, 0, 0);
}
template <int N>
static __device__ __forceinline__ void wait_async() {
#if defined(__has_builtin) && __has_builtin(__builtin_amdgcn_s_wait_asynccnt)
    __builtin_amdgcn_s_wait_asynccnt((short)N);
#else
    asm volatile("s_wait_asynccnt %0" ::"i"(N));
#endif
}
#endif

static __device__ __forceinline__ v8f v8f_zero() {
    v8f z = {0.f, 0.f, 0.f, 0.f, 0.f, 0.f, 0.f, 0.f};
    return z;
}

// fp32 -> bf16 hi/lo split (hi = rne(x), lo = rne(x - hi)); hi+lo ~ x to 2^-16.
static __device__ __forceinline__ void split2(float x, __bf16* hi, __bf16* lo) {
    __bf16 h = (__bf16)x;
    *hi = h;
    *lo = (__bf16)(x - (float)h);
}

union Frag { v16bf v; uint4 q[2]; };

// A-operand fragment (16x32 bf16): lane<16 -> K {kbase..kbase+7, kbase+16..kbase+23}
// per ISA "16-bit A-Matrix 16x32" table; caller passes kbase = half*8 (+32*chunk).
static __device__ __forceinline__ v16bf ld_frag_a(const __bf16* rowp, int kbase) {
    Frag f;
    f.q[0] = *(const uint4*)(rowp + kbase);
    f.q[1] = *(const uint4*)(rowp + kbase + 16);
    return f.v;
}

// B-operand fragment (32x16 bf16): lane<16 -> K {kbase..kbase+7, kbase+16..kbase+23}
// (same K order as A); caller passes kbase = half*8 (+32*chunk).
static __device__ __forceinline__ v16bf ld_frag_b(const __bf16* rowp, int kbase) {
    Frag f;
    f.q[0] = *(const uint4*)(rowp + kbase);
    f.q[1] = *(const uint4*)(rowp + kbase + 16);
    return f.v;
}

// D = (Ah+Al)x(Bh+Bl) + C, dropping the lo*lo term.
static __device__ __forceinline__ v8f wmma3(v16bf ah, v16bf al, v16bf bh, v16bf bl, v8f c) {
    c = WMMA_BF16(false, ah, false, bh, (short)0, c, false, false);
    c = WMMA_BF16(false, ah, false, bl, (short)0, c, false, false);
    c = WMMA_BF16(false, al, false, bh, (short)0, c, false, false);
    return c;
}

// ---------------------------------------------------------------------------
// GEMM: C[M,N] = A[M,K] @ B[K,N] + bias[N].  M,N,K multiples of 128/128/32.
// Workgroup tile 128x128, BK=32, 8 waves: 4 (M) x 2 (N), wave tile 32x64.
// Global loads for tile k+1 are issued before computing tile k (register
// double buffer) so HBM latency overlaps the WMMA phase.
// ---------------------------------------------------------------------------
__launch_bounds__(256)
__global__ void gemm_bias_wmma(const float* __restrict__ A,
                               const float* __restrict__ B,
                               const float* __restrict__ bias,
                               float* __restrict__ C,
                               int M, int N, int K) {
    constexpr int BM = 128, BN = 128, BK = 32, PAD = 8;
    __shared__ __bf16 As_hi[BM][BK + PAD];
    __shared__ __bf16 As_lo[BM][BK + PAD];
    __shared__ __bf16 Bs_hi[BN][BK + PAD];   // transposed: [n][k]
    __shared__ __bf16 Bs_lo[BN][BK + PAD];

    const int tid  = threadIdx.x;
    const int wave = tid >> 5;
    const int lane = tid & 31;
    const int l16  = lane & 15;
    const int half = lane >> 4;
    const int wm   = wave & 3;            // wave position in M (0..3) -> 32 rows
    const int wn   = wave >> 2;           // wave position in N (0..1) -> 64 cols
    const int m0   = blockIdx.y * BM;
    const int n0   = blockIdx.x * BN;

    v8f acc[2][4];
#pragma unroll
    for (int mt = 0; mt < 2; ++mt)
#pragma unroll
        for (int nt = 0; nt < 4; ++nt) acc[mt][nt] = v8f_zero();

    const int aRow = tid >> 3;            // 0..31
    const int aCol = (tid & 7) << 2;      // 0,4,...,28
    const int bKr  = tid >> 5;            // 0..7
    const int bNc  = (tid & 31) << 2;     // 0,4,...,124

    auto loadTiles = [&](int k0, float4* ar, float4* br) {
#pragma unroll
        for (int p = 0; p < 4; ++p)
            ar[p] = *(const float4*)(A + (size_t)(m0 + p * 32 + aRow) * K + (k0 + aCol));
#pragma unroll
        for (int p = 0; p < 4; ++p)
            br[p] = *(const float4*)(B + (size_t)(k0 + p * 8 + bKr) * N + (n0 + bNc));
    };

    float4 aCur[4], bCur[4];
    loadTiles(0, aCur, bCur);

    for (int k0 = 0; k0 < K; k0 += BK) {
        __syncthreads();
        // Stage current tile into LDS with bf16 hi/lo conversion.
#pragma unroll
        for (int p = 0; p < 4; ++p) {
            const int row = p * 32 + aRow;
            split2(aCur[p].x, &As_hi[row][aCol + 0], &As_lo[row][aCol + 0]);
            split2(aCur[p].y, &As_hi[row][aCol + 1], &As_lo[row][aCol + 1]);
            split2(aCur[p].z, &As_hi[row][aCol + 2], &As_lo[row][aCol + 2]);
            split2(aCur[p].w, &As_hi[row][aCol + 3], &As_lo[row][aCol + 3]);
        }
#pragma unroll
        for (int p = 0; p < 4; ++p) {
            const int kr = p * 8 + bKr;
            split2(bCur[p].x, &Bs_hi[bNc + 0][kr], &Bs_lo[bNc + 0][kr]);
            split2(bCur[p].y, &Bs_hi[bNc + 1][kr], &Bs_lo[bNc + 1][kr]);
            split2(bCur[p].z, &Bs_hi[bNc + 2][kr], &Bs_lo[bNc + 2][kr]);
            split2(bCur[p].w, &Bs_hi[bNc + 3][kr], &Bs_lo[bNc + 3][kr]);
        }
        // Issue next tile's global loads before computing (latency hiding).
        float4 aNxt[4] = {}, bNxt[4] = {};
        if (k0 + BK < K) loadTiles(k0 + BK, aNxt, bNxt);
        __syncthreads();

        v16bf ah[2], al[2], bh[4], bl[4];
#pragma unroll
        for (int mt = 0; mt < 2; ++mt) {
            const int row = wm * 32 + mt * 16 + l16;
            ah[mt] = ld_frag_a(&As_hi[row][0], half * 8);
            al[mt] = ld_frag_a(&As_lo[row][0], half * 8);
        }
#pragma unroll
        for (int nt = 0; nt < 4; ++nt) {
            const int col = wn * 64 + nt * 16 + l16;
            bh[nt] = ld_frag_b(&Bs_hi[col][0], half * 8);
            bl[nt] = ld_frag_b(&Bs_lo[col][0], half * 8);
        }
#pragma unroll
        for (int mt = 0; mt < 2; ++mt)
#pragma unroll
            for (int nt = 0; nt < 4; ++nt)
                acc[mt][nt] = wmma3(ah[mt], al[mt], bh[nt], bl[nt], acc[mt][nt]);

#pragma unroll
        for (int p = 0; p < 4; ++p) { aCur[p] = aNxt[p]; bCur[p] = bNxt[p]; }
    }

    // Epilogue: C layout VGPR r -> (M = r + 8*half, N = l16) within each tile.
#pragma unroll
    for (int mt = 0; mt < 2; ++mt)
#pragma unroll
        for (int nt = 0; nt < 4; ++nt) {
            const int col = n0 + wn * 64 + nt * 16 + l16;
            const float bv = bias[col];
#pragma unroll
            for (int r = 0; r < 8; ++r) {
                const int row = m0 + wm * 32 + mt * 16 + r + half * 8;
                C[(size_t)row * N + col] = acc[mt][nt][r] + bv;
            }
        }
}

// ---------------------------------------------------------------------------
// Flash attention: per WG 128 q-rows of one (b,h); 8 waves x 16 q-rows.
// K-blocks of 32 staged in LDS (K row-major, V transposed), online softmax.
// qkv layout: [B*T, 3072] with q|k|v at col offsets 0|1024|2048, head at h*64.
// K/V staging: async global->LDS double buffer when available, else a
// register-pipelined direct path.
// ---------------------------------------------------------------------------
__launch_bounds__(256)
__global__ void attn_wmma(const float* __restrict__ qkv, float* __restrict__ out) {
    constexpr int T = 2048, HD = 64, KB = 32, PAD = 8;
    __shared__ __bf16 Ks_hi[KB][HD + PAD];       // [k][d]
    __shared__ __bf16 Ks_lo[KB][HD + PAD];
    __shared__ __bf16 Vs_hi[HD][KB + PAD];       // transposed: [d][k]
    __shared__ __bf16 Vs_lo[HD][KB + PAD];
    __shared__ __bf16 Ps_hi[8][16][KB + PAD];    // per-wave P scratch [qrow][k]
    __shared__ __bf16 Ps_lo[8][16][KB + PAD];
#if HAVE_ASYNC_LDS
    __shared__ float Kraw[2][KB][HD];            // raw fp32 async staging
    __shared__ float Vraw[2][KB][HD];
#endif

    const int tid  = threadIdx.x;
    const int wave = tid >> 5;
    const int lane = tid & 31;
    const int l16  = lane & 15;
    const int half = lane >> 4;
    const int q0   = blockIdx.x * 128;
    const int bh   = blockIdx.y;
    const int b    = bh >> 4;
    const int h    = bh & 15;
    const size_t qkvBase = (size_t)b * T * 3072 + (size_t)h * HD;
    const int q0w  = q0 + wave * 16;

    // Preload Q fragments (scaled by 1/sqrt(64)), A-operand layout, hi/lo split.
    v16bf qh[2], ql[2];
    {
        const int qrow = q0w + l16;
        const float* qp = qkv + qkvBase + (size_t)qrow * 3072;
#pragma unroll
        for (int c = 0; c < 2; ++c) {
            Frag H, L;
            const int d0 = c * 32 + half * 8;
#pragma unroll
            for (int j = 0; j < 8; ++j) {
                __bf16 h0, l0, h1, l1;
                split2(qp[d0 + j] * 0.125f, &h0, &l0);
                split2(qp[d0 + 16 + j] * 0.125f, &h1, &l1);
                H.v[j] = h0;     L.v[j] = l0;
                H.v[8 + j] = h1; L.v[8 + j] = l1;
            }
            qh[c] = H.v; ql[c] = L.v;
        }
    }

    v8f o[4];
#pragma unroll
    for (int dt = 0; dt < 4; ++dt) o[dt] = v8f_zero();
    float mrow[8], lrow[8];
#pragma unroll
    for (int r = 0; r < 8; ++r) { mrow[r] = -1e30f; lrow[r] = 0.f; }

    const int ldR = tid >> 4;            // 0..15
    const int ldC = (tid & 15) << 2;     // 0..60
    const int kend = q0 + 128;           // causal: only blocks with kb <= q0+127

    // Convert a staged fp32 K/V block (in registers) into bf16 hi/lo LDS tiles.
    auto stageKV = [&](const float4* kq, const float4* vq) {
#pragma unroll
        for (int p = 0; p < 2; ++p) {
            const int kr = p * 16 + ldR;
            split2(kq[p].x, &Ks_hi[kr][ldC + 0], &Ks_lo[kr][ldC + 0]);
            split2(kq[p].y, &Ks_hi[kr][ldC + 1], &Ks_lo[kr][ldC + 1]);
            split2(kq[p].z, &Ks_hi[kr][ldC + 2], &Ks_lo[kr][ldC + 2]);
            split2(kq[p].w, &Ks_hi[kr][ldC + 3], &Ks_lo[kr][ldC + 3]);
            split2(vq[p].x, &Vs_hi[ldC + 0][kr], &Vs_lo[ldC + 0][kr]);
            split2(vq[p].y, &Vs_hi[ldC + 1][kr], &Vs_lo[ldC + 1][kr]);
            split2(vq[p].z, &Vs_hi[ldC + 2][kr], &Vs_lo[ldC + 2][kr]);
            split2(vq[p].w, &Vs_hi[ldC + 3][kr], &Vs_lo[ldC + 3][kr]);
        }
    };

#if HAVE_ASYNC_LDS
    // Each wave issues 4 b128 async copies per tile: chunk = (wave*4+j)*32+lane
    // covers 1024 x 16B = K (8KB) then V (8KB).
    auto issueKV = [&](int kb, int buf) {
#pragma unroll
        for (int j = 0; j < 4; ++j) {
            const int c   = (wave * 4 + j) * 32 + lane;  // 0..1023
            const int isV = c >> 9;                      // uniform per instr
            const int cc  = c & 511;
            const int row = cc >> 4;
            const int col = (cc & 15) << 2;
            const float* g = qkv + qkvBase + 1024 + (size_t)isV * 1024 +
                             (size_t)(kb + row) * 3072 + col;
            float* l = isV ? &Vraw[buf][row][col] : &Kraw[buf][row][col];
            async_cp16(g, l);
        }
    };
    issueKV(0, 0);
    int buf = 0;
#else
    auto loadKV = [&](int kb, float4* kq, float4* vq) {
#pragma unroll
        for (int p = 0; p < 2; ++p) {
            const float* kp = qkv + qkvBase + 1024 + (size_t)(kb + p * 16 + ldR) * 3072 + ldC;
            kq[p] = *(const float4*)kp;
            vq[p] = *(const float4*)(kp + 1024);
        }
    };
    float4 kCur[2], vCur[2];
    loadKV(0, kCur, vCur);
#endif

    for (int kb = 0; kb < kend; kb += KB) {
        const bool more = (kb + KB < kend);  // block-uniform
#if HAVE_ASYNC_LDS
        if (more) {                          // prefetch next tile
            issueKV(kb + KB, buf ^ 1);
            wait_async<4>();                 // current tile resident, next in flight
        } else {
            wait_async<0>();
        }
        __syncthreads();                     // all waves' copies done; prev compute done
        {
            float4 kq[2], vq[2];
#pragma unroll
            for (int p = 0; p < 2; ++p) {
                kq[p] = *(const float4*)&Kraw[buf][p * 16 + ldR][ldC];
                vq[p] = *(const float4*)&Vraw[buf][p * 16 + ldR][ldC];
            }
            stageKV(kq, vq);
        }
        buf ^= 1;
#else
        __syncthreads();                     // prev compute done reading bf16 tiles
        stageKV(kCur, vCur);
        float4 kNxt[2] = {}, vNxt[2] = {};
        if (more) loadKV(kb + KB, kNxt, vNxt);
#endif
        __syncthreads();                     // bf16 tiles ready for all waves

        if (kb <= q0w + 15) {                // wave-uniform causal skip
            // S = Q @ K^T : 2 n-tiles of 16 k-cols, K-dim = 64 (2 chunks).
            v8f s[2];
#pragma unroll
            for (int nt = 0; nt < 2; ++nt) {
                v8f accv = v8f_zero();
                const __bf16* krh = &Ks_hi[nt * 16 + l16][0];
                const __bf16* krl = &Ks_lo[nt * 16 + l16][0];
#pragma unroll
                for (int c = 0; c < 2; ++c) {
                    v16bf bhf = ld_frag_b(krh, c * 32 + half * 8);
                    v16bf blf = ld_frag_b(krl, c * 32 + half * 8);
                    accv = wmma3(qh[c], ql[c], bhf, blf, accv);
                }
                s[nt] = accv;
            }

            // Causal mask + online softmax (row stats per r, lane-half = +8).
            float mnew[8];
#pragma unroll
            for (int r = 0; r < 8; ++r) {
                const int qg = q0w + r + half * 8;
                float mx = -1e30f;
#pragma unroll
                for (int nt = 0; nt < 2; ++nt) {
                    const int kg = kb + nt * 16 + l16;
                    if (kg > qg) s[nt][r] = -1e30f;
                    mx = fmaxf(mx, s[nt][r]);
                }
                mx = fmaxf(mx, __shfl_xor(mx, 1, 32));
                mx = fmaxf(mx, __shfl_xor(mx, 2, 32));
                mx = fmaxf(mx, __shfl_xor(mx, 4, 32));
                mx = fmaxf(mx, __shfl_xor(mx, 8, 32));
                mnew[r] = fmaxf(mrow[r], mx);
            }
#pragma unroll
            for (int r = 0; r < 8; ++r) {
                const float a = __expf(mrow[r] - mnew[r]);
                float rs = 0.f;
#pragma unroll
                for (int nt = 0; nt < 2; ++nt) {
                    const float pv = __expf(s[nt][r] - mnew[r]);
                    s[nt][r] = pv;
                    rs += pv;
                }
                rs += __shfl_xor(rs, 1, 32);
                rs += __shfl_xor(rs, 2, 32);
                rs += __shfl_xor(rs, 4, 32);
                rs += __shfl_xor(rs, 8, 32);
                lrow[r] = lrow[r] * a + rs;
                mrow[r] = mnew[r];
#pragma unroll
                for (int dt = 0; dt < 4; ++dt) o[dt][r] *= a;
            }

            // Transpose P (C layout -> A layout) through wave-private LDS.
#pragma unroll
            for (int nt = 0; nt < 2; ++nt)
#pragma unroll
                for (int r = 0; r < 8; ++r) {
                    __bf16 ph, pl;
                    split2(s[nt][r], &ph, &pl);
                    Ps_hi[wave][r + half * 8][nt * 16 + l16] = ph;
                    Ps_lo[wave][r + half * 8][nt * 16 + l16] = pl;
                }

            // O += P @ V  (K-dim = 32, B-operand from transposed V).
            const v16bf pah = ld_frag_a(&Ps_hi[wave][l16][0], half * 8);
            const v16bf pal = ld_frag_a(&Ps_lo[wave][l16][0], half * 8);
#pragma unroll
            for (int dt = 0; dt < 4; ++dt) {
                const v16bf vhf = ld_frag_b(&Vs_hi[dt * 16 + l16][0], half * 8);
                const v16bf vlf = ld_frag_b(&Vs_lo[dt * 16 + l16][0], half * 8);
                o[dt] = wmma3(pah, pal, vhf, vlf, o[dt]);
            }
        }
#if !HAVE_ASYNC_LDS
#pragma unroll
        for (int p = 0; p < 2; ++p) { kCur[p] = kNxt[p]; vCur[p] = vNxt[p]; }
#endif
    }

    // Normalize and write to [B,T,D] layout for the output projection GEMM.
#pragma unroll
    for (int r = 0; r < 8; ++r) {
        const int qg = q0w + r + half * 8;
        const float inv = 1.0f / lrow[r];
        float* op = out + ((size_t)b * T + qg) * 1024 + h * HD;
#pragma unroll
        for (int dt = 0; dt < 4; ++dt)
            op[dt * 16 + l16] = o[dt][r] * inv;
    }
}

// ---------------------------------------------------------------------------
// Launch: qkv GEMM -> flash attention -> output projection GEMM.
// d_ws: [8192*3072] fp32 qkv  +  [8192*1024] fp32 attention output (128 MB).
// ---------------------------------------------------------------------------
extern "C" void kernel_launch(void* const* d_in, const int* in_sizes, int n_in,
                              void* d_out, int out_size, void* d_ws, size_t ws_size,
                              hipStream_t stream) {
    const float* x     = (const float*)d_in[0];  // [4,2048,1024]
    const float* W_qkv = (const float*)d_in[1];  // [1024,3072]
    const float* b_qkv = (const float*)d_in[2];  // [3072]
    const float* W_out = (const float*)d_in[3];  // [1024,1024]
    const float* b_out = (const float*)d_in[4];  // [1024]
    float* outp = (float*)d_out;                 // [4,2048,1024]

    float* qkv_ws  = (float*)d_ws;                       // 8192 x 3072
    float* attn_ws = qkv_ws + (size_t)8192 * 3072;       // 8192 x 1024

    const int M = 8192, D = 1024, N3 = 3072;

    gemm_bias_wmma<<<dim3(N3 / 128, M / 128), 256, 0, stream>>>(
        x, W_qkv, b_qkv, qkv_ws, M, N3, D);

    attn_wmma<<<dim3(2048 / 128, 4 * 16), 256, 0, stream>>>(qkv_ws, attn_ws);

    gemm_bias_wmma<<<dim3(D / 128, M / 128), 256, 0, stream>>>(
        attn_ws, W_out, b_out, outp, M, D, D);
}
